// BezierToImageLayer_36197984371365
// MI455X (gfx1250) — matrix operand order repaired and checked by reading, not compiled, not run on hardware
//
#include <hip/hip_runtime.h>
#include <hip/hip_bf16.h>

typedef __attribute__((ext_vector_type(16))) _Float16 v16h;
typedef __attribute__((ext_vector_type(8)))  _Float16 v8h;
typedef __attribute__((ext_vector_type(2)))  _Float16 v2h;
typedef __attribute__((ext_vector_type(8)))  float    v8f;
typedef __attribute__((ext_vector_type(2)))  float    v2f;

#define NS   30      // samples per curve
#define WIMG 60      // image width/height
#define LC   160     // curves per sample
#define KPAD 32      // padded K per curve (NS=30 -> 32)
#define MPAD 64      // padded M/N (WIMG=60 -> 64)

// exp(-d^2/ALPHA) = exp2(d^2 * (-1/ALPHA)*log2(e)) ; ALPHA = 2e-4
#define EXP2_SCALE (-7213.4752f)   // -5000 * 1.4426950408889634

// result[b,i,j] = min(1, sum_l sum_n gx[b,l,i,n] * gy[b,l,j,n])
// One 256-thread block (8 wave32) per batch element b.
__global__ __launch_bounds__(256)
void bezier_splat_wmma(const float* __restrict__ x, float* __restrict__ out)
{
    __shared__ __align__(16) float sX[LC][NS];
    __shared__ __align__(16) float sY[LC][NS];
    __shared__ __align__(16) _Float16 sGx[2][MPAD][KPAD];
    __shared__ __align__(16) _Float16 sGy[2][MPAD][KPAD];

    const int tid  = threadIdx.x;
    const int b    = blockIdx.x;
    const int lane = tid & 31;
    const int wv   = tid >> 5;

    // ---- zero tile padding once; per-curve writes only touch [0..59][0..29]
    {
        uint32_t* p0 = (uint32_t*)&sGx[0][0][0];
        uint32_t* p1 = (uint32_t*)&sGy[0][0][0];
        for (int i = tid; i < (2 * MPAD * KPAD) / 2; i += 256) {
            p0[i] = 0u;
            p1[i] = 0u;
        }
    }

    // ---- phase 0: all curve sample points for this batch into LDS
    const float* cp = x + (size_t)b * (LC * 8);
    for (int p = tid; p < LC * NS; p += 256) {
        const int l = p / NS;
        const int n = p - l * NS;
        float t = (float)n * (1.0f / 30.0f);
        t = 2.0f * t * t * t - 3.0f * t * t + 2.0f * t;   // warped parameter
        const float tb = 1.0f - t;
        const float t2 = t * t;
        const float t3 = t2 * t;
        const float w0 = t3;
        const float w1 = 3.0f * (t2 - t3);
        const float w2 = 3.0f * (t3 - 2.0f * t2 + t);
        const float w3 = tb * tb * tb;
        const float4* c4 = (const float4*)(cp + l * 8);
        const float4 cA = c4[0];   // c0x c0y c1x c1y
        const float4 cB = c4[1];   // c2x c2y c3x c3y
        sX[l][n] = w0 * cA.x + w1 * cA.z + w2 * cB.x + w3 * cB.z;
        sY[l][n] = w0 * cA.y + w1 * cA.w + w2 * cB.y + w3 * cB.w;
    }
    __syncthreads();

    // ---- hoist all per-slot fill indexing out of the curve loop.
    // 1800 pair-slots (900 gx + 900 gy), 256 threads -> 8 slots/thread,
    // slots 0..6 always valid, slot 7 only for tid < 8 (dummy -> slot 0 addrs).
    #define NSLOT 8
    float            bposA[NSLOT];   // board position for this slot's row
    const float*     srcA[NSLOT];    // &s{X,Y}[0][n0]  (advance by l*NS per curve)
    _Float16*        dstA[NSLOT];    // &sG{x,y}[0][row][n0] (advance by buf*MPAD*KPAD)
    #pragma unroll
    for (int k = 0; k < NSLOT; ++k) {
        int idx = tid + k * 256;
        if (idx >= 1800) idx = 0;            // dummy (only slot 7, tid >= 8)
        const int isY = (idx >= 900) ? 1 : 0;
        const int r   = isY ? (idx - 900) : idx;
        const int row = r / 15;
        const int pc  = r - row * 15;
        const int n0  = pc * 2;
        bposA[k] = (float)row * (1.0f / 60.0f);
        srcA[k]  = (isY ? &sY[0][0] : &sX[0][0]) + n0;
        dstA[k]  = (isY ? &sGy[0][0][0] : &sGx[0][0][0]) + row * KPAD + n0;
    }
    const bool slot7ok = (tid + 7 * 256) < 1800;

    // ---- per-wave output tile assignment: wave wv owns tiles (mi,nj0),(mi,nj0+1)
    const int mi   = wv >> 1;          // 0..3
    const int nj0  = (wv & 1) * 2;     // 0 or 2
    v8f c0 = {};
    v8f c1 = {};

    // fragment addressing (contiguous k = 16*(lane/16) + s, same order for A and B)
    const int arow  = mi * 16 + (lane & 15);
    const int kaddA = (lane & 16) ? 16 : 0;    // A: lanes 16-31 hold K=16..31
    const int kbB   = (lane & 16) ? 16 : 0;    // B: lanes 16-31 hold K=16..31
    const int bcol0 = nj0 * 16 + (lane & 15);
    const int bcol1 = bcol0 + 16;

    for (int l = 0; l < LC; ++l) {
        const int buf = l & 1;
        const int dofs = buf * (MPAD * KPAD);  // half-elements
        const int sofs = l * NS;               // float elements

        // --- phase A: batch all point-pair loads (back-to-back ds_load_b64)
        v2f P[NSLOT];
        #pragma unroll
        for (int k = 0; k < NSLOT; ++k) {
            P[k] = *(const v2f*)(srcA[k] + sofs);
        }

        // --- phase B: splat, pack f16 pairs, store to tile
        #pragma unroll
        for (int k = 0; k < NSLOT; ++k) {
            const float d0 = bposA[k] - P[k].x;
            const float d1 = bposA[k] - P[k].y;
            const float g0 = __builtin_amdgcn_exp2f(d0 * d0 * EXP2_SCALE);
            const float g1 = __builtin_amdgcn_exp2f(d1 * d1 * EXP2_SCALE);
            v2h pk;
            pk.x = (_Float16)g0;
            pk.y = (_Float16)g1;
            if (k < NSLOT - 1 || slot7ok) {
                *(v2h*)(dstA[k] + dofs) = pk;
            }
        }
        __syncthreads();

        // A fragment: row arow, halves [kaddA..kaddA+15] (32 contiguous bytes)
        union { v16h v; v8h h[2]; } au;
        au.h[0] = *(const v8h*)&sGx[buf][arow][kaddA];
        au.h[1] = *(const v8h*)&sGx[buf][arow][kaddA + 8];

        // B fragments: col bcolX, halves [kbB..kbB+15] (32 contiguous bytes)
        union { v16h v; v8h h[2]; } b0u, b1u;
        b0u.h[0] = *(const v8h*)&sGy[buf][bcol0][kbB];
        b0u.h[1] = *(const v8h*)&sGy[buf][bcol0][kbB + 8];
        b1u.h[0] = *(const v8h*)&sGy[buf][bcol1][kbB];
        b1u.h[1] = *(const v8h*)&sGy[buf][bcol1][kbB + 8];

        c0 = __builtin_amdgcn_wmma_f32_16x16x32_f16(
                 false, au.v, false, b0u.v, (short)0, c0, false, false);
        c1 = __builtin_amdgcn_wmma_f32_16x16x32_f16(
                 false, au.v, false, b1u.v, (short)0, c1, false, false);
    }

    // ---- epilogue: clamp and scatter per C/D layout
    const int ibase = mi * 16 + ((lane & 16) ? 8 : 0);
    const int j0    = nj0 * 16 + (lane & 15);
    const int j1    = j0 + 16;
    float* outb = out + (size_t)b * (WIMG * WIMG);
    #pragma unroll
    for (int r = 0; r < 8; ++r) {
        const int i = ibase + r;
        if (i < WIMG) {
            if (j0 < WIMG) outb[i * WIMG + j0] = fminf(c0[r], 1.0f);
            if (j1 < WIMG) outb[i * WIMG + j1] = fminf(c1[r], 1.0f);
        }
    }
}

extern "C" void kernel_launch(void* const* d_in, const int* in_sizes, int n_in,
                              void* d_out, int out_size, void* d_ws, size_t ws_size,
                              hipStream_t stream)
{
    const float* x = (const float*)d_in[0];
    float* out = (float*)d_out;
    const int B = in_sizes[0] / (LC * 8);   // 256
    bezier_splat_wmma<<<B, 256, 0, stream>>>(x, out);
}
